// SepConvTemp3D_2448131359031
// MI455X (gfx1250) — matrix operand order repaired and checked by reading, not compiled, not run on hardware
//
#include <hip/hip_runtime.h>
#include <hip/hip_bf16.h>
#include <stdint.h>

// Problem geometry (fixed by the reference): T=8, C=1024, H=W=64.
#define C_DIM   1024
#define T_DIM   8
#define HW_DIM  64
#define NVOX    (T_DIM * HW_DIM * HW_DIM)     // 32768 voxels per channel
#define TOT     (C_DIM * NVOX)                // 33,554,432 elements

// GEMM: M = C_out = 1024, N = NVOX = 32768, K = C_in = 1024
#define GEMM_M  1024
#define GEMM_N  32768
#define GEMM_K  1024
#define BM      128     // block tile M (4 waves x 32)
#define BN      256     // block tile N (2 waves x 128)
#define BK      32      // K per WMMA step

typedef __attribute__((ext_vector_type(16))) __bf16 v16bf;
typedef __attribute__((ext_vector_type(8)))  float  v8f;

union Frag { v16bf v; uint4 q[2]; };

__device__ __forceinline__ void split_bf16(float x, __bf16& hi, __bf16& lo) {
    __bf16 h = (__bf16)x;                 // RNE
    hi = h;
    lo = (__bf16)(x - (float)h);
}

// ---------------------------------------------------------------------------
// Kernel 0: split the 1024x1024 pointwise weight into bf16 hi/lo planes.
// Layout stays [o][c] row-major (K-contiguous) -> direct A-fragment loads.
// ---------------------------------------------------------------------------
__global__ void pw_split_kernel(const float* __restrict__ pw,
                                __bf16* __restrict__ hi, __bf16* __restrict__ lo) {
    int i = blockIdx.x * 256 + threadIdx.x;   // exactly 1,048,576 threads
    split_bf16(pw[i], hi[i], lo[i]);
}

// ---------------------------------------------------------------------------
// Kernel 1: depthwise 3x3x3 conv (shared kernel), fused with:
//   - fp32 -> bf16 hi/lo split
//   - LDS transpose so the output is stored K-major: yT[voxel][channel].
// ---------------------------------------------------------------------------
#define CB    64          // channels per block
#define CPAD  72          // padded LDS row (bf16) -> 144B rows, 16B aligned
__global__ __launch_bounds__(256) void dw_split_tr_kernel(
    const float* __restrict__ x, const float* __restrict__ w27,
    __bf16* __restrict__ yThi, __bf16* __restrict__ yTlo) {

    __shared__ __align__(16) __bf16 YsHi[HW_DIM][CPAD];
    __shared__ __align__(16) __bf16 YsLo[HW_DIM][CPAD];

    const int tid  = threadIdx.x;
    const int vrow = blockIdx.x;          // 0..511 : (t,h) row
    const int cb   = blockIdx.y * CB;     // channel block base
    const int tv   = vrow >> 6;           // 0..7
    const int hv   = vrow & 63;           // 0..63

    float k[27];
#pragma unroll
    for (int j = 0; j < 27; ++j) k[j] = w27[j];   // uniform -> scalar loads

    // ---- phase 1: stencil for 64 channels x 64 w, staged in LDS ----
    {
        const int wv  = tid & 63;
        const int cl0 = tid >> 6;          // 0..3
#pragma unroll 4
        for (int i = 0; i < 16; ++i) {
            const int cl = cl0 + 4 * i;    // 0..63
            const float* base = x + (size_t)(cb + cl) * NVOX;
            float acc = 0.f;
#pragma unroll
            for (int dt = -1; dt <= 1; ++dt) {
                int tt = tv + dt;
                if ((unsigned)tt >= T_DIM) continue;
#pragma unroll
                for (int dh = -1; dh <= 1; ++dh) {
                    int hh = hv + dh;
                    if ((unsigned)hh >= HW_DIM) continue;
#pragma unroll
                    for (int dw = -1; dw <= 1; ++dw) {
                        int ww = wv + dw;
                        if ((unsigned)ww >= HW_DIM) continue;
                        acc += k[(dt + 1) * 9 + (dh + 1) * 3 + (dw + 1)]
                             * base[tt * 4096 + hh * 64 + ww];
                    }
                }
            }
            __bf16 h, l;
            split_bf16(acc, h, l);
            YsHi[wv][cl] = h;
            YsLo[wv][cl] = l;
        }
    }
    __syncthreads();

    // ---- phase 2: transposed vectorized store: yT[v][c], 128B per v row ----
    {
        const int chunk = tid & 7;         // 8 chunks of 8 channels = 64
        const int vv    = tid >> 3;        // 0..31, handles vv and vv+32
#pragma unroll
        for (int s = 0; s < 2; ++s) {
            const int v  = vv + s * 32;
            const size_t go = (size_t)(vrow * 64 + v) * GEMM_K + cb + chunk * 8;
            *(uint4*)(yThi + go) = *(const uint4*)(&YsHi[v][chunk * 8]);
            *(uint4*)(yTlo + go) = *(const uint4*)(&YsLo[v][chunk * 8]);
        }
    }
}

// ---------------------------------------------------------------------------
// Kernel 2: pointwise GEMM  out[1024][32768] = A[1024][1024] * yT^T
// Split-bf16 WMMA (ah*bh + ah*bl + al*bh), fp32 accumulators.
// LDS-free. All per-lane address math is a loop-invariant 32-bit byte
// offset; only uniform (SGPR) base pointers advance per k-step, and tile
// selection lives in the 24-bit instruction offset -> no VALU in the loop,
// no WMMA->VALU WAR hazard nops.
// ---------------------------------------------------------------------------
__global__ __launch_bounds__(256) void pw_gemm_wmma(
    const __bf16* __restrict__ Ahi_, const __bf16* __restrict__ Alo_,  // [1024][1024]
    const __bf16* __restrict__ Bhi_, const __bf16* __restrict__ Blo_,  // [32768][1024] K-major
    float* __restrict__ out) {

    const int tid  = threadIdx.x;
    const int lane = tid & 31;
    const int half = (lane >> 4) & 1;
    const int l16  = lane & 15;
    const int wave = tid >> 5;
    const int wm   = wave & 3;     // 4 waves along M
    const int wn   = wave >> 2;    // 2 waves along N
    const int m0   = blockIdx.y * BM + wm * 32;
    const int n0   = blockIdx.x * BN + wn * 128;

    // Loop-invariant per-lane byte offsets (fit in 32 bits: buffers are 64 MiB).
    //  A 16x32 bf16 frag: lanes 0-15 K{0..7,16..23}, lanes 16-31 K{8..15,24..31}
    //  B 32x16 bf16 frag: lanes 0-15 K{0..7,16..23}, lanes 16-31 K{8..15,24..31} (N = lane%16)
    const uint32_t aOff = (uint32_t)(((m0 + l16) * GEMM_K + half * 8)  * (int)sizeof(__bf16));
    const uint32_t bOff = (uint32_t)(((n0 + l16) * GEMM_K + half * 8) * (int)sizeof(__bf16));

    const char* pAh = (const char*)Ahi_;
    const char* pAl = (const char*)Alo_;
    const char* pBh = (const char*)Bhi_;
    const char* pBl = (const char*)Blo_;

    v8f acc[2][8] = {};

    for (int k0 = 0; k0 < GEMM_K; k0 += BK) {
        Frag aHi[2], aLo[2];
#pragma unroll
        for (int mt = 0; mt < 2; ++mt) {
            // row step mt*16 -> mt*16*1024*2B = 32768B: folds into the 24-bit ioffset
            aHi[mt].q[0] = *(const uint4*)(pAh + aOff + mt * 32768 +  0);
            aHi[mt].q[1] = *(const uint4*)(pAh + aOff + mt * 32768 + 32);
            aLo[mt].q[0] = *(const uint4*)(pAl + aOff + mt * 32768 +  0);
            aLo[mt].q[1] = *(const uint4*)(pAl + aOff + mt * 32768 + 32);
        }

#pragma unroll
        for (int nt = 0; nt < 8; ++nt) {
            Frag bHi, bLo;
            bHi.q[0] = *(const uint4*)(pBh + bOff + nt * 32768 +  0);
            bHi.q[1] = *(const uint4*)(pBh + bOff + nt * 32768 + 32);
            bLo.q[0] = *(const uint4*)(pBl + bOff + nt * 32768 +  0);
            bLo.q[1] = *(const uint4*)(pBl + bOff + nt * 32768 + 32);
#pragma unroll
            for (int mt = 0; mt < 2; ++mt) {
                acc[mt][nt] = __builtin_amdgcn_wmma_f32_16x16x32_bf16(
                    false, aHi[mt].v, false, bHi.v, (short)0, acc[mt][nt], false, false);
                acc[mt][nt] = __builtin_amdgcn_wmma_f32_16x16x32_bf16(
                    false, aHi[mt].v, false, bLo.v, (short)0, acc[mt][nt], false, false);
                acc[mt][nt] = __builtin_amdgcn_wmma_f32_16x16x32_bf16(
                    false, aLo[mt].v, false, bHi.v, (short)0, acc[mt][nt], false, false);
            }
        }

        // advance only the uniform (scalar) bases: BK elements = 64 bytes
        pAh += BK * sizeof(__bf16);
        pAl += BK * sizeof(__bf16);
        pBh += BK * sizeof(__bf16);
        pBl += BK * sizeof(__bf16);
    }

    // D 16x16 f32 layout: lanes 0-15 (N=lane): M=r; lanes 16-31: M=8+r
#pragma unroll
    for (int mt = 0; mt < 2; ++mt) {
#pragma unroll
        for (int nt = 0; nt < 8; ++nt) {
            const int col = n0 + nt * 16 + l16;
#pragma unroll
            for (int r = 0; r < 8; ++r) {
                const int row = m0 + mt * 16 + half * 8 + r;
                out[(size_t)row * GEMM_N + col] = acc[mt][nt][r];
            }
        }
    }
}

// ---------------------------------------------------------------------------
extern "C" void kernel_launch(void* const* d_in, const int* in_sizes, int n_in,
                              void* d_out, int out_size, void* d_ws, size_t ws_size,
                              hipStream_t stream) {
    const float* features = (const float*)d_in[0];   // [T,C,H,W] == [C,T,H,W] raw
    const float* dw_w     = (const float*)d_in[1];   // 27 taps
    const float* pw_w     = (const float*)d_in[2];   // [1024,1024]
    float* out            = (float*)d_out;

    char* ws = (char*)d_ws;
    __bf16* yThi = (__bf16*)ws;                                   // 64 MiB  [32768][1024]
    __bf16* yTlo = yThi + (size_t)TOT;                            // 64 MiB
    __bf16* pwhi = (__bf16*)(ws + 2ull * TOT * sizeof(__bf16));   //  2 MiB
    __bf16* pwlo = pwhi + (size_t)GEMM_M * GEMM_K;                //  2 MiB

    // 0) split pointwise weight to bf16 hi/lo
    pw_split_kernel<<<(GEMM_M * GEMM_K) / 256, 256, 0, stream>>>(pw_w, pwhi, pwlo);

    // 1) depthwise conv + split + K-major transpose of the intermediate
    dim3 g1(NVOX / HW_DIM, C_DIM / CB);   // 512 x 16
    dw_split_tr_kernel<<<g1, 256, 0, stream>>>(features, dw_w, yThi, yTlo);

    // 2) pointwise channel-mixing GEMM on the WMMA pipe (LDS-free)
    dim3 g2(GEMM_N / BN, GEMM_M / BM);    // 128 x 8
    pw_gemm_wmma<<<g2, 256, 0, stream>>>(pwhi, pwlo, yThi, yTlo, out);
}
